// MoELayer_56856777065247
// MI455X (gfx1250) — matrix operand order repaired and checked by reading, not compiled, not run on hardware
//
#include <hip/hip_runtime.h>

#define B_ 4
#define S_ 1024
#define D_ 768
#define F_ 3072
#define E_ 8
#define T_ (B_ * S_)

#if __has_builtin(__builtin_amdgcn_global_load_async_to_lds_b128) && \
    __has_builtin(__builtin_amdgcn_s_wait_asynccnt)
#define HAVE_ASYNC_LDS 1
#else
#define HAVE_ASYNC_LDS 0
#endif

typedef __attribute__((ext_vector_type(16))) __bf16 v16bf;
typedef __attribute__((ext_vector_type(8)))  float  v8f;
typedef int v4i_ __attribute__((vector_size(16)));   // pointee type of async-LDS builtins

union BfVec { v16bf v; unsigned short u[16]; uint4 q[2]; };

__device__ __forceinline__ unsigned short f32_to_bf16(float f) {
    union { float f; unsigned int u; } x;
    x.f = f;
    unsigned int lsb = (x.u >> 16) & 1u;
    unsigned int r = x.u + 0x7fffu + lsb;   // round to nearest even
    return (unsigned short)(r >> 16);
}

// A fragment (16x32 bf16, MxK) from LDS row-major [16][ldx] ushort.
// Lane layout per ISA: lanes 0-15 -> M=lane, K in {0..7,16..23}; lanes 16-31 -> +8.
// Two contiguous 8-element runs -> 2 x ds_load_b128.
__device__ __forceinline__ v16bf load_a_frag(const unsigned short* sm, int lane,
                                             int kb, int ldx) {
    const int m  = lane & 15;
    const int kh = (lane >> 4) << 3;                 // 0 or 8
    const unsigned short* p = sm + m * ldx + kb + kh;
    BfVec a;
    a.q[0] = *(const uint4*)p;                       // K = kh .. kh+7
    a.q[1] = *(const uint4*)(p + 16);                // K = kh+16 .. kh+23
    return a.v;
}

// B fragment (32x16 bf16, KxN) from K-contiguous (pre-transposed) weights.
// g0 points at (k=0, n=0) of the fragment; column n starts at g0 + n*ldk.
// lanes 0-15: K={0..7,16..23}, lanes 16-31: K={8..15,24..31} -> 2 x global_load_b128.
__device__ __forceinline__ v16bf load_b_frag(const unsigned short* g0, int lane,
                                             int ldk) {
    const int n     = lane & 15;
    const int kbase = (lane >> 4) << 3;              // 0 or 8
    const unsigned short* p = g0 + (long)n * ldk + kbase;
    BfVec b;
    b.q[0] = *(const uint4*)p;                       // K = kbase .. +7
    b.q[1] = *(const uint4*)(p + 16);                // K = kbase+16 .. +23
    return b.v;
}

__device__ __forceinline__ v8f wmma_bf16(v16bf a, v16bf b, v8f c) {
    return __builtin_amdgcn_wmma_f32_16x16x32_bf16(false, a, false, b,
                                                   (short)0, c, false, false);
}

__global__ void moe_init(int* counts) {
    if (threadIdx.x < E_) counts[threadIdx.x] = 0;
}

__global__ void moe_cvt_x(const float* __restrict__ x,
                          unsigned short* __restrict__ xb) {
    long i = (long)blockIdx.x * blockDim.x + threadIdx.x;
    const long nTot = (long)T_ * D_;
    const long stride = (long)gridDim.x * blockDim.x;
    for (; i < nTot; i += stride) xb[i] = f32_to_bf16(x[i]);
}

// Transpose+convert one [R][C] f32 matrix per expert into [C][R] bf16.
// 32x32 tiles staged through LDS so global reads AND writes stay coalesced.
__global__ __launch_bounds__(256) void moe_cvt_t(const float* __restrict__ in,
                                                 unsigned short* __restrict__ outb,
                                                 int R, int C) {
    __shared__ unsigned short tile[32][33];
    const int tilesC = C >> 5;
    const int tilesPerMat = (R >> 5) * tilesC;
    const int e  = blockIdx.x / tilesPerMat;
    const int t  = blockIdx.x % tilesPerMat;
    const int tr = t / tilesC;
    const int tc = t % tilesC;
    const float* src = in + (long)e * R * C;
    unsigned short* dst = outb + (long)e * R * C;    // out is [C][R]
    const int r = threadIdx.x >> 5;                  // 0..7
    const int c = threadIdx.x & 31;
#pragma unroll
    for (int j = 0; j < 4; ++j) {
        const int rr = r + 8 * j;
        tile[rr][c] = f32_to_bf16(src[(long)(tr * 32 + rr) * C + tc * 32 + c]);
    }
    __syncthreads();
#pragma unroll
    for (int j = 0; j < 4; ++j) {
        const int rr = r + 8 * j;                    // output row = input col
        dst[(long)(tc * 32 + rr) * R + tr * 32 + c] = tile[c][rr];
    }
}

__global__ void moe_router(const float* __restrict__ x, const float* __restrict__ wr,
                           float* __restrict__ logits_out, float* __restrict__ idx_out,
                           float* __restrict__ probs, int* __restrict__ counts,
                           int* __restrict__ lists) {
    const int t = blockIdx.x * blockDim.x + threadIdx.x;
    if (t >= T_) return;
    float acc[E_] = {};
    const float* xr = x + (long)t * D_;
    for (int d = 0; d < D_; ++d) {
        const float xv = xr[d];
        const float* w = wr + d * E_;
#pragma unroll
        for (int e = 0; e < E_; ++e) acc[e] = fmaf(xv, w[e], acc[e]);
    }
    float lmax = acc[0];
    int best = 0;
#pragma unroll
    for (int e = 1; e < E_; ++e)
        if (acc[e] > lmax) { lmax = acc[e]; best = e; }
    float sum = 0.f;
#pragma unroll
    for (int e = 0; e < E_; ++e) sum += __expf(acc[e] - lmax);
#pragma unroll
    for (int e = 0; e < E_; ++e) logits_out[(long)t * E_ + e] = acc[e];
    idx_out[t] = (float)best;
    probs[t] = 1.0f / sum;                    // softmax prob of argmax
    const int pos = atomicAdd(&counts[best], 1);
    lists[best * T_ + pos] = t;
}

#define LDX (D_ + 8)   // 776 elems: rows stay 16B aligned
#define LDH 136
#define MT  32         // tokens per tile

__global__ __launch_bounds__(256) void moe_ffn(
    const unsigned short* __restrict__ xb,     // [T][D] bf16
    const unsigned short* __restrict__ w1t,    // [E][F][D] bf16 (K-contig)
    const unsigned short* __restrict__ w2t,    // [E][D][F] bf16 (K-contig)
    const float* __restrict__ b1, const float* __restrict__ b2,
    const float* __restrict__ probs, const int* __restrict__ counts,
    const int* __restrict__ lists, float* __restrict__ out) {
    __shared__ int   tok[MT];
    __shared__ float scale[MT];
    __shared__ __align__(16) unsigned short xs[MT * LDX];
    __shared__ __align__(16) unsigned short hs[MT * LDH];

    const int e     = blockIdx.x >> 7;         // 128 tiles per expert
    const int tile  = blockIdx.x & 127;
    const int count = counts[e];
    const int m0    = tile * MT;
    if (m0 >= count) return;                   // uniform across block

    const int tid  = threadIdx.x;
    const int lane = tid & 31;
    const int w    = tid >> 5;                 // wave 0..7
    const int n    = lane & 15;

    if (tid < MT) {
        const int idx = m0 + tid;
        const int tkn = lists[e * T_ + (idx < count ? idx : count - 1)];
        tok[tid]   = tkn;
        scale[tid] = probs[tkn];
    }
    __syncthreads();

    {   // stage x tile: bf16 copy at b128 granularity
        const int row = tid >> 3;              // 0..31
        const int c0  = tid & 7;
        const unsigned short* src = xb + (long)tok[row] * D_;
        unsigned short* dst = xs + row * LDX;
#if HAVE_ASYNC_LDS
        // gfx1250 async copy: global -> LDS without touching VGPRs (ASYNCcnt)
#pragma unroll
        for (int j = 0; j < 12; ++j) {
            const int col = (c0 + 8 * j) * 8;
            __builtin_amdgcn_global_load_async_to_lds_b128(
                (__attribute__((address_space(1))) v4i_*)(src + col),
                (__attribute__((address_space(3))) v4i_*)(dst + col),
                0, 0);
        }
        __builtin_amdgcn_s_wait_asynccnt(0);
#else
#pragma unroll
        for (int j = 0; j < 12; ++j) {
            const int col = (c0 + 8 * j) * 8;
            *(uint4*)(dst + col) = *(const uint4*)(src + col);
        }
#endif
    }
    __syncthreads();

    v8f oacc[2][6];
#pragma unroll
    for (int mh = 0; mh < 2; ++mh)
#pragma unroll
        for (int j = 0; j < 6; ++j) oacc[mh][j] = {};

    const unsigned short* w1te = w1t + (long)e * F_ * D_;
    const unsigned short* w2te = w2t + (long)e * D_ * F_;

    for (int fc = 0; fc < F_; fc += 128) {
        // ---- GEMM1: wave w -> h[:, fc+16w .. +16), two M-halves share B ----
        {
            const int fcol0 = fc + 16 * w;
            const unsigned short* bcol = w1te + (long)fcol0 * D_;
            v8f h0 = {}, h1 = {};
            for (int kb = 0; kb < D_; kb += 32) {
                v16bf a0 = load_a_frag(xs, lane, kb, LDX);
                v16bf a1 = load_a_frag(xs + 16 * LDX, lane, kb, LDX);
                v16bf b  = load_b_frag(bcol + kb, lane, D_);
                __builtin_prefetch(bcol + kb + 256, 0, 1);   // global_prefetch_b8
                h0 = wmma_bf16(a0, b, h0);
                h1 = wmma_bf16(a1, b, h1);
            }
            const float bias = b1[e * F_ + fcol0 + n];
            unsigned short* hdst = hs + (16 * w + n);
#pragma unroll
            for (int r = 0; r < 8; ++r) {
                const int mlo = (lane < 16) ? r : (r + 8);
                float v0 = h0[r] + bias; v0 = v0 > 0.f ? v0 : 0.f;   // ReLU
                float v1 = h1[r] + bias; v1 = v1 > 0.f ? v1 : 0.f;
                hdst[mlo * LDH]        = f32_to_bf16(v0);
                hdst[(mlo + 16) * LDH] = f32_to_bf16(v1);
            }
        }
        __syncthreads();
        // ---- GEMM2: wave w owns out cols [96w, 96w+96), K chunk = 128 ----
#pragma unroll
        for (int kb = 0; kb < 128; kb += 32) {
            v16bf a0 = load_a_frag(hs, lane, kb, LDH);
            v16bf a1 = load_a_frag(hs + 16 * LDH, lane, kb, LDH);
#pragma unroll
            for (int j = 0; j < 6; ++j) {
                const int col0 = 16 * (w * 6 + j);
                v16bf b = load_b_frag(w2te + (long)col0 * F_ + fc + kb, lane, F_);
                oacc[0][j] = wmma_bf16(a0, b, oacc[0][j]);
                oacc[1][j] = wmma_bf16(a1, b, oacc[1][j]);
            }
        }
        __syncthreads();
    }

    // ---- writeout: (acc + b2) * top1_prob, masked to valid rows ----
#pragma unroll
    for (int j = 0; j < 6; ++j) {
        const int dcol = 16 * (w * 6 + j) + n;
        const float bias2 = b2[e * D_ + dcol];
#pragma unroll
        for (int mh = 0; mh < 2; ++mh) {
#pragma unroll
            for (int r = 0; r < 8; ++r) {
                const int m = mh * 16 + ((lane < 16) ? r : (r + 8));
                if (m0 + m < count)
                    out[(long)tok[m] * D_ + dcol] =
                        (oacc[mh][j][r] + bias2) * scale[m];
            }
        }
    }
}

extern "C" void kernel_launch(void* const* d_in, const int* in_sizes, int n_in,
                              void* d_out, int out_size, void* d_ws, size_t ws_size,
                              hipStream_t stream) {
    const float* x  = (const float*)d_in[0];
    const float* wr = (const float*)d_in[1];
    const float* w1 = (const float*)d_in[2];
    const float* b1 = (const float*)d_in[3];
    const float* w2 = (const float*)d_in[4];
    const float* b2 = (const float*)d_in[5];

    float* out    = (float*)d_out;                 // [T, D]
    float* logits = out + (long)T_ * D_;           // [T, E]
    float* idxout = logits + (long)T_ * E_;        // [T]

    char* ws = (char*)d_ws;
    float* probs  = (float*)ws;                                  // T floats
    int*   counts = (int*)(ws + 16384);                          // E ints
    int*   lists  = (int*)(ws + 16384 + 256);                    // E*T ints
    unsigned short* xb  = (unsigned short*)(ws + (1l << 20));    // T*D bf16
    unsigned short* w1t = (unsigned short*)(ws + (8l << 20));    // E*F*D bf16
    unsigned short* w2t = w1t + (size_t)E_ * F_ * D_;            // E*D*F bf16

    moe_init<<<1, 32, 0, stream>>>(counts);
    moe_cvt_x<<<3072, 256, 0, stream>>>(x, xb);
    moe_cvt_t<<<E_ * (D_ / 32) * (F_ / 32), 256, 0, stream>>>(w1, w1t, D_, F_);
    moe_cvt_t<<<E_ * (F_ / 32) * (D_ / 32), 256, 0, stream>>>(w2, w2t, F_, D_);
    moe_router<<<T_ / 256, 256, 0, stream>>>(x, wr, logits, idxout, probs,
                                             counts, lists);
    moe_ffn<<<E_ * 128, 256, 0, stream>>>(xb, w1t, w2t, b1, b2, probs, counts,
                                          lists, out);
}
